// Graph_Conv_13142599926064
// MI455X (gfx1250) — compile-verified, matrix operand order checked by reading, not run on hardware
//
#include <hip/hip_runtime.h>
#include <hip/hip_bf16.h>

typedef __attribute__((ext_vector_type(16))) _Float16 v16h;
typedef __attribute__((ext_vector_type(8)))  _Float16 v8h;
typedef __attribute__((ext_vector_type(8)))  float    v8f;

#define NCH   10000
#define NDV   1000000
#define BATCH 8192
#define KN    64
#define EMB   16
#define DVD   20
#define IN1   32     // channel self feature dim
#define IN2   132    // device feature dim
#define IN2P  160    // padded to 5*32 for f16 WMMA K
#define H1    64
#define H2    64
#define ATTN  84
#define AK    96     // H2 + IN1 (already 3*32)
#define ANP   96     // attn out padded (84 -> 96)

// ---------------------------------------------------------------------------
// WMMA fragment loader for 16-bit A/B operands (ISA 7.12.2 layout).
// Row-major source [rows x ld] in halfs; fragment covers rows [0,16) x K [0,32).
// Lane L: row = L&15; K-halves: VGPR0..3 = K in [h*8, h*8+8), VGPR4..7 = +16,
// where h = L>>4.  Two 16B loads per lane.
// ---------------------------------------------------------------------------
__device__ inline v16h load_frag16(const _Float16* __restrict__ base, int ld) {
  int lane = threadIdx.x & 31;
  int m    = lane & 15;
  int half = lane >> 4;
  const v8h* p0 = (const v8h*)(base + m * ld + half * 8);
  const v8h* p1 = (const v8h*)(base + m * ld + half * 8 + 16);
  v8h lo = *p0;
  v8h hi = *p1;
  return __builtin_shufflevector(lo, hi, 0, 1, 2, 3, 4, 5, 6, 7,
                                 8, 9, 10, 11, 12, 13, 14, 15);
}

// ---------------------------------------------------------------------------
// K0: transpose + f16-convert + zero-pad the three weight matrices.
//   waggt  [64][160] <- W_agg [132][64]   (K padded 132->160)
//   wselft [64][32]  <- W_self[32][64]
//   wattnt [96][96]  <- W_attn[96][84]    (N padded 84->96 with zeros)
// ---------------------------------------------------------------------------
__global__ void k_prep_weights(const float* __restrict__ Wagg,
                               const float* __restrict__ Wself,
                               const float* __restrict__ Wattn,
                               _Float16* __restrict__ waggt,
                               _Float16* __restrict__ wselft,
                               _Float16* __restrict__ wattnt) {
  int i = blockIdx.x * blockDim.x + threadIdx.x;
  if (i < H2 * IN2P) {
    int n = i / IN2P, k = i % IN2P;
    waggt[i] = (_Float16)((k < IN2) ? Wagg[k * H2 + n] : 0.0f);
  } else if (i < H2 * IN2P + H1 * IN1) {
    int j = i - H2 * IN2P;
    int n = j / IN1, k = j % IN1;
    wselft[j] = (_Float16)Wself[k * H1 + n];
  } else if (i < H2 * IN2P + H1 * IN1 + ANP * AK) {
    int j = i - H2 * IN2P - H1 * IN1;
    int n = j / AK, k = j % AK;
    wattnt[j] = (_Float16)((n < ATTN) ? Wattn[k * ATTN + n] : 0.0f);
  }
}

// ---------------------------------------------------------------------------
// K1: build channel self features ch_v[B,32] (f32 + f16 copy) and scatter the
// f16 copy into the attention concat buffer cols [64,96) for both sides.
// ---------------------------------------------------------------------------
__global__ void k_chv(const float* __restrict__ chdense,
                      const float* __restrict__ cemb,
                      const int*   __restrict__ cids,
                      const int*   __restrict__ verts,
                      float*       __restrict__ chv,
                      _Float16*    __restrict__ chvh,
                      _Float16*    __restrict__ xattn) {
  int b = blockIdx.x * 8 + (threadIdx.x >> 5);
  int d = threadIdx.x & 31;
  if (b >= BATCH) return;
  int v = verts[b];
  float val = (d < 16) ? chdense[v * 16 + d] : cemb[cids[v] * 16 + (d - 16)];
  chv[b * IN1 + d] = val;
  _Float16 hv = (_Float16)val;
  chvh[b * IN1 + d] = hv;
  xattn[b * AK + H2 + d]           = hv;
  xattn[(BATCH + b) * AK + H2 + d] = hv;
}

// ---------------------------------------------------------------------------
// K2: gather neighbor device features, masked (prefix) mean, write f16 rows
// padded to 160.  One block per (side, vertex).  160 threads = feature dims
// (132 active, 28 write pad zeros).  Neighbor ids + categorical ids staged in
// LDS so the inner loop is one global load per (dim, neighbor).
// ---------------------------------------------------------------------------
__global__ void k_agg_mean(const float* __restrict__ ddense,
                           const int*   __restrict__ dcat,
                           const float* __restrict__ t0, const float* __restrict__ t1,
                           const float* __restrict__ t2, const float* __restrict__ t3,
                           const float* __restrict__ t4, const float* __restrict__ t5,
                           const float* __restrict__ t6,
                           const int* __restrict__ bot_n, const int* __restrict__ nor_n,
                           const int* __restrict__ bot_c, const int* __restrict__ nor_c,
                           _Float16* __restrict__ xagg) {
  __shared__ int nidx[KN];
  __shared__ int cats[KN * 7];
  int row = blockIdx.x;             // 0..2B-1
  int s   = row / BATCH;
  int b   = row % BATCH;
  const int* nb = (s == 0 ? bot_n : nor_n) + b * KN;
  int count = (s == 0 ? bot_c[b] : nor_c[b]);
  if (count < 1) count = 1;

  int t = threadIdx.x;
  if (t < KN) nidx[t] = nb[t];
  __syncthreads();
  for (int i = t; i < KN * 7; i += blockDim.x)
    cats[i] = dcat[nidx[i / 7] * 7 + (i % 7)];
  __syncthreads();

  float acc = 0.0f;
  int d = t;
  if (d < DVD) {
    for (int k = 0; k < count; k++)
      acc += ddense[nidx[k] * DVD + d];
  } else if (d < IN2) {
    int e = (d - DVD) >> 4;
    int c = (d - DVD) & 15;
    const float* tbl;
    switch (e) {
      case 0: tbl = t0; break; case 1: tbl = t1; break;
      case 2: tbl = t2; break; case 3: tbl = t3; break;
      case 4: tbl = t4; break; case 5: tbl = t5; break;
      default: tbl = t6; break;
    }
    for (int k = 0; k < count; k++)
      acc += tbl[cats[k * 7 + e] * EMB + c];
  }
  if (d < IN2P)
    xagg[row * IN2P + d] = (_Float16)(d < IN2 ? acc / (float)count : 0.0f);
}

// ---------------------------------------------------------------------------
// K3: H[16384,64] = xagg[16384,160] @ Waggt^T + b_agg   (f16 WMMA, f32 acc)
// Epilogue also writes the f16 result into attn concat cols [0,64).
// Block: 128 threads = 4 waves, wave w owns N-tile w; block owns one M-tile.
// ---------------------------------------------------------------------------
__global__ void k_gemm_agg(const _Float16* __restrict__ xagg,
                           const _Float16* __restrict__ waggt,
                           const float*    __restrict__ bagg,
                           float*          __restrict__ Hb,
                           _Float16*       __restrict__ xattn) {
  int wave = threadIdx.x >> 5, lane = threadIdx.x & 31;
  int m0 = blockIdx.x * 16;
  int n0 = wave * 16;
  v8f acc = {};
#pragma unroll
  for (int kk = 0; kk < IN2P / 32; kk++) {
    v16h a  = load_frag16(xagg  + m0 * IN2P + kk * 32, IN2P);
    v16h bf = load_frag16(waggt + n0 * IN2P + kk * 32, IN2P);
    acc = __builtin_amdgcn_wmma_f32_16x16x32_f16(false, a, false, bf,
                                                 (short)0, acc, false, false);
  }
  int half = lane >> 4, n = lane & 15;
  int col = n0 + n;
  float bias = bagg[col];
#pragma unroll
  for (int r = 0; r < 8; r++) {
    int row = m0 + r + 8 * half;
    float v = acc[r] + bias;
    Hb[row * H2 + col]    = v;
    xattn[row * AK + col] = (_Float16)v;
  }
}

// ---------------------------------------------------------------------------
// K4: S[16384,96] = relu(xattn[16384,96] @ Wattnt^T + b_attn)
// Block: 192 threads = 6 waves (N tiles 0..5, cols 84..95 are inert padding).
// ---------------------------------------------------------------------------
__global__ void k_gemm_attn(const _Float16* __restrict__ xattn,
                            const _Float16* __restrict__ wattnt,
                            const float*    __restrict__ battn,
                            float*          __restrict__ S) {
  int wave = threadIdx.x >> 5, lane = threadIdx.x & 31;
  int m0 = blockIdx.x * 16;
  int n0 = wave * 16;
  v8f acc = {};
#pragma unroll
  for (int kk = 0; kk < AK / 32; kk++) {
    v16h a  = load_frag16(xattn  + m0 * AK + kk * 32, AK);
    v16h bf = load_frag16(wattnt + n0 * AK + kk * 32, AK);
    acc = __builtin_amdgcn_wmma_f32_16x16x32_f16(false, a, false, bf,
                                                 (short)0, acc, false, false);
  }
  int half = lane >> 4, n = lane & 15;
  int col = n0 + n;
  float bias = (col < ATTN) ? battn[col] : 0.0f;
#pragma unroll
  for (int r = 0; r < 8; r++) {
    int row = m0 + r + 8 * half;
    float v = acc[r] + bias;
    S[row * ANP + col] = v > 0.0f ? v : 0.0f;
  }
}

// ---------------------------------------------------------------------------
// K5: selfH[8192,64] = chvh[8192,32] @ Wselft^T + b_self  (single K step)
// ---------------------------------------------------------------------------
__global__ void k_gemm_self(const _Float16* __restrict__ chvh,
                            const _Float16* __restrict__ wselft,
                            const float*    __restrict__ bself,
                            float*          __restrict__ selfH) {
  int wave = threadIdx.x >> 5, lane = threadIdx.x & 31;
  int m0 = blockIdx.x * 16;
  int n0 = wave * 16;
  v16h a  = load_frag16(chvh   + m0 * IN1, IN1);
  v16h bf = load_frag16(wselft + n0 * IN1, IN1);
  v8f acc = {};
  acc = __builtin_amdgcn_wmma_f32_16x16x32_f16(false, a, false, bf,
                                               (short)0, acc, false, false);
  int half = lane >> 4, n = lane & 15;
  int col = n0 + n;
  float bias = bself[col];
#pragma unroll
  for (int r = 0; r < 8; r++)
    selfH[(m0 + r + 8 * half) * H1 + col] = acc[r] + bias;
}

// ---------------------------------------------------------------------------
// K6: per vertex — two 84-dots vs W_attn2, 2-way softmax, convex combine of
// bot/normal H rows, relu, concat with relu(selfH).  out[B,128].
// ---------------------------------------------------------------------------
__global__ void k_final(const float* __restrict__ S,
                        const float* __restrict__ W2,
                        const float* __restrict__ b2,
                        const float* __restrict__ Hb,
                        const float* __restrict__ selfH,
                        float*       __restrict__ out) {
  __shared__ float red[128];
  __shared__ float aw[2];
  int b = blockIdx.x, t = threadIdx.x;

  float pb = (t < ATTN) ? S[b * ANP + t] * W2[t] : 0.0f;
  red[t] = pb; __syncthreads();
  for (int s = 64; s > 0; s >>= 1) { if (t < s) red[t] += red[t + s]; __syncthreads(); }
  float sb = red[0] + b2[0]; __syncthreads();

  float pn = (t < ATTN) ? S[(BATCH + b) * ANP + t] * W2[t] : 0.0f;
  red[t] = pn; __syncthreads();
  for (int s = 64; s > 0; s >>= 1) { if (t < s) red[t] += red[t + s]; __syncthreads(); }
  float sn = red[0] + b2[0]; __syncthreads();

  if (t == 0) {
    float m  = sb > sn ? sb : sn;
    float e0 = __expf(sb - m), e1 = __expf(sn - m);
    float inv = 1.0f / (e0 + e1);
    aw[0] = e0 * inv; aw[1] = e1 * inv;
  }
  __syncthreads();
  float a0 = aw[0], a1 = aw[1];

  if (t < H2) {
    float v = a0 * Hb[b * H2 + t] + a1 * Hb[(BATCH + b) * H2 + t];
    out[b * (H2 + H1) + t] = v > 0.0f ? v : 0.0f;
  } else {
    int j = t - H2;
    float v = selfH[b * H1 + j];
    out[b * (H2 + H1) + H2 + j] = v > 0.0f ? v : 0.0f;
  }
}

// ---------------------------------------------------------------------------
extern "C" void kernel_launch(void* const* d_in, const int* in_sizes, int n_in,
                              void* d_out, int out_size, void* d_ws, size_t ws_size,
                              hipStream_t stream) {
  const float* channel_dense = (const float*)d_in[0];
  const float* device_dense  = (const float*)d_in[1];
  const float* channel_emb   = (const float*)d_in[2];
  const float* lang_emb      = (const float*)d_in[3];
  const float* plat_emb      = (const float*)d_in[4];
  const float* os_emb        = (const float*)d_in[5];
  const float* country_emb   = (const float*)d_in[6];
  const float* carrier_emb   = (const float*)d_in[7];
  const float* brand_emb     = (const float*)d_in[8];
  const float* plat_os_emb   = (const float*)d_in[9];
  const float* W_agg   = (const float*)d_in[10];
  const float* b_agg   = (const float*)d_in[11];
  const float* W_self  = (const float*)d_in[12];
  const float* b_self  = (const float*)d_in[13];
  const float* W_attn  = (const float*)d_in[14];
  const float* b_attn  = (const float*)d_in[15];
  const float* W_attn2 = (const float*)d_in[16];
  const float* b_attn2 = (const float*)d_in[17];
  const int* channel_ids   = (const int*)d_in[18];
  const int* device_cat    = (const int*)d_in[19];
  const int* vertices      = (const int*)d_in[20];
  const int* bot_neibrs    = (const int*)d_in[21];
  const int* normal_neibrs = (const int*)d_in[22];
  const int* bot_counts    = (const int*)d_in[23];
  const int* normal_counts = (const int*)d_in[24];
  float* out = (float*)d_out;

  // workspace carve-out (256B aligned slices, ~23 MB total)
  char* ws = (char*)d_ws;
  size_t off = 0;
  auto carve = [&](size_t bytes) {
    void* p = ws + off;
    off = (off + bytes + 255) & ~(size_t)255;
    return p;
  };
  _Float16* xagg   = (_Float16*)carve((size_t)2 * BATCH * IN2P * sizeof(_Float16));
  _Float16* waggt  = (_Float16*)carve((size_t)H2 * IN2P * sizeof(_Float16));
  _Float16* wselft = (_Float16*)carve((size_t)H1 * IN1 * sizeof(_Float16));
  _Float16* wattnt = (_Float16*)carve((size_t)ANP * AK * sizeof(_Float16));
  float*    chv    = (float*)   carve((size_t)BATCH * IN1 * sizeof(float));
  _Float16* chvh   = (_Float16*)carve((size_t)BATCH * IN1 * sizeof(_Float16));
  _Float16* xattn  = (_Float16*)carve((size_t)2 * BATCH * AK * sizeof(_Float16));
  float*    Hb     = (float*)   carve((size_t)2 * BATCH * H2 * sizeof(float));
  float*    S      = (float*)   carve((size_t)2 * BATCH * ANP * sizeof(float));
  float*    selfH  = (float*)   carve((size_t)BATCH * H1 * sizeof(float));

  // 1) weight prep (tiny)
  k_prep_weights<<<84, 256, 0, stream>>>(W_agg, W_self, W_attn, waggt, wselft, wattnt);
  // 2) channel self features (also fills attn-concat cols 64..95)
  k_chv<<<BATCH / 8, 256, 0, stream>>>(channel_dense, channel_emb, channel_ids,
                                       vertices, chv, chvh, xattn);
  // 3) neighbor gather + masked mean (dominant, L2-resident gather)
  k_agg_mean<<<2 * BATCH, IN2P, 0, stream>>>(device_dense, device_cat,
      lang_emb, plat_emb, os_emb, country_emb, carrier_emb, brand_emb, plat_os_emb,
      bot_neibrs, normal_neibrs, bot_counts, normal_counts, xagg);
  // 4) aggregation GEMM (WMMA f16): 16384x160 @ 160x64
  k_gemm_agg<<<2 * BATCH / 16, 128, 0, stream>>>(xagg, waggt, b_agg, Hb, xattn);
  // 5) self GEMM (WMMA f16): 8192x32 @ 32x64
  k_gemm_self<<<BATCH / 16, 128, 0, stream>>>(chvh, wselft, b_self, selfH);
  // 6) attention GEMM (WMMA f16): 16384x96 @ 96x96(pad of 84) + bias + relu
  k_gemm_attn<<<2 * BATCH / 16, 192, 0, stream>>>(xattn, wattnt, b_attn, S);
  // 7) scores, softmax, combine, relu, concat
  k_final<<<BATCH, 128, 0, stream>>>(S, W_attn2, b_attn2, Hb, selfH, out);
}
